// TrajectoryDecoder_61581241090073
// MI455X (gfx1250) — compile-verified, matrix operand order checked by reading, not run on hardware
//
#include <hip/hip_runtime.h>
#include <hip/hip_bf16.h>
#include <stdint.h>

// Problem constants (from reference)
#define N_ROWS 32768
#define P_DIM  2
#define EMB_D  128
#define H_DIM  256
#define M_DIM  2
#define T_DIM  8
#define KTOT   640     // EMB + H (aug=384) + H (recurrent=256)
#define GATES  1024    // 4*H
#define KTILES 20      // 640 / 32

typedef __attribute__((ext_vector_type(16))) _Float16 v16h;
typedef __attribute__((ext_vector_type(8)))  _Float16 v8h;
typedef __attribute__((ext_vector_type(8)))  float    v8f;

__device__ __forceinline__ float wave_sum32(float v) {
#pragma unroll
  for (int o = 16; o > 0; o >>= 1) v += __shfl_xor(v, o, 32);
  return v;
}
__device__ __forceinline__ float sigmoidf(float x) { return 1.0f / (1.0f + __expf(-x)); }

// ---------------------------------------------------------------------------
// K1: repack [Wk;Wr] (f32, 640x1024 row-major) into f16 B-fragment-major
// layout for v_wmma_f32_16x16x32_f16. For each (ktile, col) we store 32 f16
// in fragment order: group0 = K {0..7,16..23}, group1 = K {8..15,24..31}.
// Group id tid&1, col (tid>>1)&1023, ktile tid>>11  -> 32 bytes per tid.
// ---------------------------------------------------------------------------
__global__ void convert_weights_kernel(const float* __restrict__ Wk,
                                       const float* __restrict__ Wr,
                                       _Float16* __restrict__ Bbuf) {
  int tid = blockIdx.x * blockDim.x + threadIdx.x;
  if (tid >= KTILES * GATES * 2) return;
  int g  = tid & 1;
  int n  = (tid >> 1) & (GATES - 1);
  int kt = tid >> 11;
  _Float16* dst = Bbuf + (size_t)tid * 16;
#pragma unroll
  for (int i = 0; i < 16; ++i) {
    int kk = (i < 8) ? (g * 8 + i) : (16 + g * 8 + (i - 8));
    int k  = kt * 32 + kk;
    float w = (k < 384) ? Wk[(size_t)k * GATES + n]
                        : Wr[(size_t)(k - 384) * GATES + n];
    dst[i] = (_Float16)w;
  }
}

// ---------------------------------------------------------------------------
// K2: focal attention + embedding + f16 activation packing.
// One wave per batch row. The 16x256 f32 context tile is DMA'd straight into
// a per-wave LDS slice with global_load_async_to_lds_b128 (ASYNCcnt path, no
// VGPR round-trip); both the logits pass and the attended-sum pass then read
// it from LDS. Writes Wft (N,2,8) and xcat[n] = [emb|attended|h0] (f16, 640).
// ---------------------------------------------------------------------------
__global__ void attention_pack_kernel(const float* __restrict__ dec_input,
                                      const float* __restrict__ h0,
                                      const float* __restrict__ context,
                                      const float* __restrict__ W_emb,
                                      const float* __restrict__ b_emb,
                                      float* __restrict__ wft_out,
                                      _Float16* __restrict__ xcat) {
  __shared__ float ctx_lds[4][16][256];          // 64 KB, one 16KB slice/wave
  const int lane = threadIdx.x & 31;
  const int wv   = threadIdx.x >> 5;
  const int n    = blockIdx.x * 4 + wv;
  float (*cl)[256] = ctx_lds[wv];

  __builtin_prefetch(W_emb, 0, 1);               // global_prefetch_b8

  // Kick off async copy of this row's 16x256 context tile (16 KB, contiguous)
  // into the wave's LDS slice: 32 iterations x 32 lanes x 16B.
  {
    const uint32_t lds_base = (uint32_t)(uintptr_t)(&ctx_lds[wv][0][0]);
    const uint64_t gbase =
        (uint64_t)(uintptr_t)(context + (size_t)n * (M_DIM * T_DIM * H_DIM));
#pragma unroll
    for (int it = 0; it < 32; ++it) {
      uint32_t off  = (uint32_t)(it * 512 + lane * 16);
      uint32_t ldsa = lds_base + off;
      uint64_t ga   = gbase + off;
      asm volatile("global_load_async_to_lds_b128 %0, %1, off"
                   :: "v"(ldsa), "v"(ga) : "memory");
    }
  }

  // query (h0 row) + its squared norm (overlaps with the async DMA)
  float q[8];
  float qq = 0.0f;
#pragma unroll
  for (int i = 0; i < 8; ++i) {
    q[i] = h0[(size_t)n * H_DIM + lane + 32 * i];
    qq += q[i] * q[i];
  }
  qq = wave_sum32(qq);
  const float qinv = rsqrtf(fmaxf(qq, 1e-12f));

  asm volatile("s_wait_asynccnt 0x0" ::: "memory");

  // logits[m,t] = <qn, cn> over the LDS-resident tile
  float logits[16];
#pragma unroll 1
  for (int idx = 0; idx < 16; ++idx) {
    float cc = 0.0f, qc = 0.0f;
#pragma unroll
    for (int i = 0; i < 8; ++i) {
      float cv = cl[idx][lane + 32 * i];
      cc += cv * cv;
      qc += q[i] * cv;
    }
    cc = wave_sum32(cc);
    qc = wave_sum32(qc);
    logits[idx] = qc * rsqrtf(fmaxf(cc, 1e-12f)) * qinv;
  }

  // softmax over t (per m), then softmax over m of the per-m max
  float at[16];
  float ml[2];
#pragma unroll
  for (int m = 0; m < 2; ++m) {
    float mx = -1e30f;
#pragma unroll
    for (int t = 0; t < 8; ++t) mx = fmaxf(mx, logits[m * 8 + t]);
    ml[m] = mx;
    float s = 0.0f;
#pragma unroll
    for (int t = 0; t < 8; ++t) {
      float e = __expf(logits[m * 8 + t] - mx);
      at[m * 8 + t] = e;
      s += e;
    }
    float inv = 1.0f / s;
#pragma unroll
    for (int t = 0; t < 8; ++t) at[m * 8 + t] *= inv;
  }
  const float mm = fmaxf(ml[0], ml[1]);
  const float e0 = __expf(ml[0] - mm);
  const float e1 = __expf(ml[1] - mm);
  const float am0 = e0 / (e0 + e1);
  const float am1 = e1 / (e0 + e1);

  float wft[16];
#pragma unroll
  for (int idx = 0; idx < 16; ++idx) wft[idx] = ((idx < 8) ? am0 : am1) * at[idx];

  if (lane < 16) {
    float v = 0.0f;
#pragma unroll
    for (int idx = 0; idx < 16; ++idx) v = (lane == idx) ? wft[idx] : v;
    wft_out[(size_t)n * 16 + lane] = v;
  }

  // attended[d] = sum_{m,t} Wft[m,t] * context[m,t,d]; pack activations f16
  _Float16* xrow = xcat + (size_t)n * KTOT;
#pragma unroll
  for (int i = 0; i < 8; ++i) {
    int d = lane + 32 * i;
    float a = 0.0f;
#pragma unroll
    for (int idx = 0; idx < 16; ++idx) a += wft[idx] * cl[idx][d];
    xrow[EMB_D + d]         = (_Float16)a;      // attended -> [128,384)
    xrow[EMB_D + H_DIM + d] = (_Float16)q[i];   // h0       -> [384,640)
  }
  const float x0 = dec_input[(size_t)n * 2 + 0];
  const float x1 = dec_input[(size_t)n * 2 + 1];
#pragma unroll
  for (int i = 0; i < 4; ++i) {
    int e = lane + 32 * i;
    float v = x0 * W_emb[e] + x1 * W_emb[EMB_D + e] + b_emb[e];
    xrow[e] = (_Float16)fmaxf(v, 0.0f);         // relu(emb) -> [0,128)
  }
}

// ---------------------------------------------------------------------------
// Byte-offset fragment loaders for K3 (single base pointer + 24-bit immediate
// offsets; keeps only two live 64-bit address pairs in the whole GEMM loop).
// ---------------------------------------------------------------------------
__device__ __forceinline__ v16h load_a_off(const char* abase, int byte_off) {
  v8h alo = *(const v8h*)(abase + byte_off);
  v8h ahi = *(const v8h*)(abase + byte_off + 32);
  v16h a;
#pragma unroll
  for (int i = 0; i < 8; ++i) { a[i] = alo[i]; a[8 + i] = ahi[i]; }
  return a;
}
__device__ __forceinline__ v16h load_b_off(const char* bbase, int byte_off) {
  return *(const v16h*)(bbase + byte_off);
}

// ---------------------------------------------------------------------------
// K3: z = xcat @ [Wk;Wr] via v_wmma_f32_16x16x32_f16 with register ping-pong
// double buffering (load ktile k+1 while the matrix pipe chews ktile k), plus
// fused LSTM gate math and the out_xy projection. One block = 16 batch rows x
// 1024 gate columns: 8 waves x 8 column tiles, 160 WMMAs/wave.
// __launch_bounds__(256, 1): allocate VGPRs for pressure (~215 needed), the
// back-to-back WMMA chain saturates the matrix pipe at low occupancy.
// ---------------------------------------------------------------------------
__global__ void __launch_bounds__(256, 1)
lstm_gemm_kernel(const _Float16* __restrict__ xcat,
                 const _Float16* __restrict__ Bbuf,
                 const float* __restrict__ b_lstm,
                 const float* __restrict__ c0,
                 const float* __restrict__ dec_input,
                 const float* __restrict__ W_out,
                 const float* __restrict__ b_out,
                 float* __restrict__ out_xy,
                 float* __restrict__ h_out,
                 float* __restrict__ c_out) {
  __shared__ float zs[16 * GATES];               // 64 KB z staging
  const int lane = threadIdx.x & 31;
  const int wv   = threadIdx.x >> 5;
  const int g    = lane >> 4;                    // K half-group (A/B frag)
  const int col  = lane & 15;                    // M row (A) / N col (B)
  const int n0   = blockIdx.x * 16;

  // Wave-constant base pointers; everything else is an immediate offset.
  // B fragment (kt, j) at bbase + kt*65536 + j*1024 bytes.
  const char* bbase = (const char*)Bbuf + (size_t)wv * 8192 + col * 64 + g * 32;
  // A fragment (kt) at abase + kt*64 (lo) / +32 (hi) bytes.
  const char* abase = (const char*)xcat + (size_t)(n0 + col) * (KTOT * 2) + g * 16;

  v8f acc[8] = {};

  // Prologue: buffer 0 <- ktile 0
  v16h a0 = load_a_off(abase, 0);
  v16h a1;
  v16h b0[8], b1[8];
#pragma unroll
  for (int j = 0; j < 8; ++j) b0[j] = load_b_off(bbase, j * 1024);

#pragma unroll 1
  for (int kt = 0; kt < KTILES; kt += 2) {
    const char* bk  = bbase + (size_t)kt * 65536;
    const char* ak  = abase + (size_t)kt * 64;

    // Load ktile kt+1 into buffer 1 (KTILES even -> always valid)
    a1 = load_a_off(ak, 64);
#pragma unroll
    for (int j = 0; j < 8; ++j) b1[j] = load_b_off(bk, 65536 + j * 1024);

    // Consume buffer 0 while buffer-1 loads are in flight
#pragma unroll
    for (int j = 0; j < 8; ++j)
      acc[j] = __builtin_amdgcn_wmma_f32_16x16x32_f16(
          false, a0, false, b0[j], (short)0, acc[j], false, false);

    // Load ktile kt+2 into buffer 0
    if (kt + 2 < KTILES) {
      a0 = load_a_off(ak, 128);
#pragma unroll
      for (int j = 0; j < 8; ++j) b0[j] = load_b_off(bk, 131072 + j * 1024);
    }

    // Consume buffer 1
#pragma unroll
    for (int j = 0; j < 8; ++j)
      acc[j] = __builtin_amdgcn_wmma_f32_16x16x32_f16(
          false, a1, false, b1[j], (short)0, acc[j], false, false);
  }

  // Spill C fragments: VGPR r, lane L -> z[row r+8*g][col]
#pragma unroll
  for (int j = 0; j < 8; ++j) {
    int cb = (wv * 8 + j) * 16 + col;
#pragma unroll
    for (int r = 0; r < 8; ++r) zs[(r + 8 * g) * GATES + cb] = acc[j][r];
  }
  __syncthreads();

  // Gate epilogue: thread tid owns hidden column j=tid for all 16 rows
  const int jj = threadIdx.x;
  float hreg[16];
#pragma unroll
  for (int r = 0; r < 16; ++r) {
    float zi = zs[r * GATES +        jj] + b_lstm[jj];
    float zf = zs[r * GATES + 256 + jj] + b_lstm[256 + jj];
    float zg = zs[r * GATES + 512 + jj] + b_lstm[512 + jj];
    float zo = zs[r * GATES + 768 + jj] + b_lstm[768 + jj];
    int n = n0 + r;
    float cn = sigmoidf(zf) * c0[(size_t)n * H_DIM + jj] + sigmoidf(zi) * tanhf(zg);
    float hn = sigmoidf(zo) * tanhf(cn);
    c_out[(size_t)n * H_DIM + jj] = cn;
    h_out[(size_t)n * H_DIM + jj] = hn;
    hreg[r] = hn;
  }
  __syncthreads();
#pragma unroll
  for (int r = 0; r < 16; ++r) zs[r * H_DIM + jj] = hreg[r];
  __syncthreads();

  // out_xy = h_new @ W_out + b_out + dec_input (256x2 per row)
  if (threadIdx.x < 32) {
    int r = threadIdx.x >> 1, p = threadIdx.x & 1;
    float s = b_out[p];
    for (int j2 = 0; j2 < 256; ++j2) s += zs[r * H_DIM + j2] * W_out[j2 * 2 + p];
    int n = n0 + r;
    out_xy[(size_t)n * 2 + p] = s + dec_input[(size_t)n * 2 + p];
  }
}

// ---------------------------------------------------------------------------
extern "C" void kernel_launch(void* const* d_in, const int* in_sizes, int n_in,
                              void* d_out, int out_size, void* d_ws, size_t ws_size,
                              hipStream_t stream) {
  const float* dec_input = (const float*)d_in[0];
  const float* h0        = (const float*)d_in[1];
  const float* c0        = (const float*)d_in[2];
  const float* context   = (const float*)d_in[3];
  const float* W_emb     = (const float*)d_in[4];
  const float* b_emb     = (const float*)d_in[5];
  const float* Wk        = (const float*)d_in[6];
  const float* Wr        = (const float*)d_in[7];
  const float* b_lstm    = (const float*)d_in[8];
  const float* W_out     = (const float*)d_in[9];
  const float* b_out     = (const float*)d_in[10];

  // Outputs concatenated flat in return order: out_xy, h_new, c_new, Wft
  float* out     = (float*)d_out;
  float* out_xy  = out;
  float* h_out   = out_xy + (size_t)N_ROWS * P_DIM;
  float* c_out   = h_out + (size_t)N_ROWS * H_DIM;
  float* wft_out = c_out + (size_t)N_ROWS * H_DIM;

  // Workspace: [ Bbuf f16 (1.31 MB) | xcat f16 (41.9 MB) ]
  _Float16* Bbuf = (_Float16*)d_ws;
  const size_t bbuf_bytes = (size_t)KTILES * GATES * 2 * 16 * sizeof(_Float16);
  _Float16* xcat = (_Float16*)((char*)d_ws + bbuf_bytes);

  convert_weights_kernel<<<(KTILES * GATES * 2 + 255) / 256, 256, 0, stream>>>(
      Wk, Wr, Bbuf);
  attention_pack_kernel<<<N_ROWS / 4, 128, 0, stream>>>(
      dec_input, h0, context, W_emb, b_emb, wft_out, xcat);
  lstm_gemm_kernel<<<N_ROWS / 16, 256, 0, stream>>>(
      xcat, Bbuf, b_lstm, c0, dec_input, W_out, b_out, out_xy, h_out, c_out);
}
